// LFSM_18932215841222
// MI455X (gfx1250) — matrix operand order repaired and checked by reading, not compiled, not run on hardware
//
#include <hip/hip_runtime.h>
#include <hip/hip_bf16.h>
#include <math.h>

// ---------------------------------------------------------------------------
// LFSM frequency-fusion forward for MI455X (gfx1250), wave32 + WMMA + TDM.
// DFTs as bf16 WMMA GEMMs (f32 accumulate) with TDM-staged A panels in LDS
// and fragment-major (swizzled) B operands; conv tower and per-pixel dynamic
// conv / gating chain are fused VALU kernels.
// ---------------------------------------------------------------------------

typedef __attribute__((ext_vector_type(16))) __bf16 bf16x16;
typedef __attribute__((ext_vector_type(8)))  __bf16 bf16x8;
typedef __attribute__((ext_vector_type(8)))  float  f32x8;
typedef __attribute__((ext_vector_type(4)))  unsigned int u32x4_;
typedef __attribute__((ext_vector_type(8)))  unsigned int u32x8_;

#define HH    512
#define WW    512
#define WFQ   257
#define HWF   (512 * 257)         // 131584
#define HWIMG (512 * 512)
#define NSTR  32                  // 16 vis + 16 ir streams
#define SWZT  (16 * 17 * 512)     // swizzled B-matrix elems: kblk*ntile*(32 lanes*16)

// ---- WMMA per-lane index helpers (contiguous k order; A and B match) -----------
// 16-bit A matrix 16x32: lane L holds row M=L&15; code's order: e -> K = 16*(L>=16)+e (not ISA table).
__device__ __forceinline__ int wmma_a_k(int lane, int e) {
  return e + ((lane >> 4) << 4);
}
// 16-bit B matrix 32x16: lane L holds col N=L&15; code's order: e -> K = 16*(L>=16)+e (not ISA table).
// Swizzled storage: elem (k,col) lives at ((k>>5)*17 + (col>>4))*512
//                   + (((k>>4)&1)<<4 | (col&15))*16 + (k&15).

__device__ __forceinline__ float sigmoidf_(float x) { return 1.0f / (1.0f + expf(-x)); }
__device__ __forceinline__ float reluf_(float x) { return x > 0.0f ? x : 0.0f; }

// ---------------------------------------------------------------------------
// Tensor Data Mover: 2-D tile Global -> LDS (CDNA5 ISA).
// LDS pad: 1 dword after every 256 dwords => 514-dword row pitch for 512-elem
// f32 rows (pitch 514 floats) / 512-elem bf16 rows (pitch 514 bf16).
// ---------------------------------------------------------------------------
__device__ __forceinline__ void lfsm_tdm_load_2d(unsigned lds_off, const void* gptr,
                                                 unsigned dsize_code,   // 1=2B, 2=4B
                                                 unsigned dim0, unsigned rows,
                                                 unsigned stride_elems) {
  unsigned long long ga = (unsigned long long)(uintptr_t)gptr;
  u32x4_ g0;
  g0[0] = 1u;                                                     // count=1
  g0[1] = (unsigned)__builtin_amdgcn_readfirstlane((int)lds_off); // lds_addr
  g0[2] = (unsigned)__builtin_amdgcn_readfirstlane((int)(unsigned)ga);
  g0[3] = ((unsigned)__builtin_amdgcn_readfirstlane((int)(unsigned)(ga >> 32)) & 0x1FFFFFFu)
        | (2u << 30);                                             // type=2 (image)
  u32x8_ g1;
  g1[0] = ((dsize_code & 3u) << 16)   // data_size
        | (1u << 20)                  // pad_enable
        | (7u << 22);                 // pad_interval = 256 dwords (pad_amount=0 -> 1 dword)
  g1[1] = (dim0 & 0xFFFFu) << 16;                       // tensor_dim0 lo
  g1[2] = (dim0 >> 16) | ((rows & 0xFFFFu) << 16);      // tensor_dim0 hi | tensor_dim1 lo
  g1[3] = (rows >> 16) | ((dim0 & 0xFFFFu) << 16);      // tensor_dim1 hi | tile_dim0
  g1[4] = rows & 0xFFFFu;                               // tile_dim1 (tile_dim2=0)
  g1[5] = stride_elems;                                 // tensor_dim0_stride lo32
  g1[6] = 0u;
  g1[7] = 0u;
  asm volatile("tensor_load_to_lds %0, %1" : : "s"(g0), "s"(g1) : "memory");
}

// ---------------------------------------------------------------------------
// Init kernels
// ---------------------------------------------------------------------------
__global__ void lfsm_zero_kernel(float* pooled, float* gsum) {
  int i = blockIdx.x * blockDim.x + threadIdx.x;
  if (i < NSTR * 32) pooled[i] = 0.0f;
  else if (i < NSTR * 32 + NSTR * 8) gsum[i - NSTR * 32] = 0.0f;
}

// Fr/Fi: row-major [512][512] column-DFT matrix (A operand, TDM-staged).
// Crs/Cis: swizzled row-rDFT matrix (B operand), zero-padded cols 257..271.
__global__ void lfsm_twiddle_kernel(__bf16* Crs, __bf16* Cis, __bf16* Fr, __bf16* Fi) {
  const float twopi = 6.283185307179586f;
  int idx = blockIdx.x * blockDim.x + threadIdx.x;
  if (idx < HWIMG) {
    int r = idx >> 9, c = idx & 511;
    float ang = -twopi * (float)((r * c) & 511) / 512.0f;
    Fr[idx] = (__bf16)cosf(ang);
    Fi[idx] = (__bf16)sinf(ang);
  }
  if (idx < SWZT) {
    int e = idx & 15, lane = (idx >> 4) & 31, t = idx >> 9;
    int nt = t % 17, kblk = t / 17;
    int k = kblk * 32 + ((lane >> 4) << 4) + e;
    int col = nt * 16 + (lane & 15);
    float vr = 0.0f, vi = 0.0f;
    if (col < WFQ) {
      float ang = -twopi * (float)((k * col) & 511) / 512.0f;
      vr = cosf(ang); vi = sinf(ang);
    }
    Crs[idx] = (__bf16)vr;
    Cis[idx] = (__bf16)vi;
  }
}

// ---------------------------------------------------------------------------
// Row rDFT: X[row,k] = sum_c img[row,c] * e^{-2pi i c k/512}.
// Block = (stream, mtile): TDM-stage 16 image rows to LDS; 8 waves cover the
// 17 N-tiles; results stored in swizzled layout for the column DFT B operand.
// ---------------------------------------------------------------------------
__global__ __launch_bounds__(256)
void lfsm_rowdft_kernel(const float* __restrict__ vis, const float* __restrict__ ir,
                        const __bf16* __restrict__ Crs, const __bf16* __restrict__ Cis,
                        __bf16* __restrict__ Xrs, __bf16* __restrict__ Xis) {
  __shared__ __align__(16) float sImg[16 * 514];   // 514-float row pitch (TDM pad)
  const int s     = blockIdx.x;
  const int mtile = blockIdx.y;
  const float* img = (s < 16) ? (vis + (size_t)s * HWIMG)
                              : (ir + (size_t)(s - 16) * HWIMG);
  if (threadIdx.x < 32) {
    lfsm_tdm_load_2d((unsigned)(uintptr_t)&sImg[0], img + (size_t)mtile * 16 * WW,
                     2u, 512u, 16u, 512u);
  }
  __builtin_amdgcn_s_wait_tensorcnt(0);
  __syncthreads();

  const int lane = threadIdx.x & 31;
  const int wv   = threadIdx.x >> 5;
  const int rowbase = (lane & 15) * 514;
  const size_t sb = (size_t)s * SWZT;

  for (int nt = wv; nt < 17; nt += 8) {
    f32x8 accR = {}, accI = {};
    for (int k0 = 0; k0 < 512; k0 += 32) {
      const int padoff = (k0 >= 256) ? 1 : 0;
      bf16x16 a;
#pragma unroll
      for (int e = 0; e < 16; ++e)
        a[e] = (__bf16)sImg[rowbase + k0 + padoff + wmma_a_k(lane, e)];

      const size_t boff = (size_t)((k0 >> 5) * 17 + nt) * 512 + lane * 16;
      const bf16x8* bpr = (const bf16x8*)(Crs + boff);
      const bf16x8* bpi = (const bf16x8*)(Cis + boff);
      __builtin_prefetch(Crs + boff + 17 * 512, 0, 1);
      bf16x8 r0 = bpr[0], r1 = bpr[1], i0 = bpi[0], i1 = bpi[1];
      bf16x16 br = __builtin_shufflevector(r0, r1, 0,1,2,3,4,5,6,7,8,9,10,11,12,13,14,15);
      bf16x16 bi = __builtin_shufflevector(i0, i1, 0,1,2,3,4,5,6,7,8,9,10,11,12,13,14,15);

      accR = __builtin_amdgcn_wmma_f32_16x16x32_bf16(false, a, false, br, (short)0, accR, false, false);
      accI = __builtin_amdgcn_wmma_f32_16x16x32_bf16(false, a, false, bi, (short)0, accI, false, false);
    }
    // Swizzled store: row0 = mtile*16 + 8*(lane>=16); 8 accs are contiguous e's.
    const int row0 = mtile * 16 + ((lane >> 4) << 3);
    const size_t idx = (size_t)((row0 >> 5) * 17 + nt) * 512
                     + (size_t)((((row0 >> 4) & 1) << 4) | (lane & 15)) * 16
                     + (row0 & 15);
    bf16x8 str, sti;
#pragma unroll
    for (int i = 0; i < 8; ++i) { str[i] = (__bf16)accR[i]; sti[i] = (__bf16)accI[i]; }
    *(bf16x8*)(Xrs + sb + idx) = str;
    *(bf16x8*)(Xis + sb + idx) = sti;
  }
}

// ---------------------------------------------------------------------------
// Column complex DFT: Y = F @ (Xr + i Xi); 4 WMMAs per k-step.
// Block = (stream, mtile): TDM-stage Fr/Fi 16-row panels to LDS.
// Emits mag -> workspace, phase -> d_out blocks 4/5.
// ---------------------------------------------------------------------------
__global__ __launch_bounds__(256)
void lfsm_coldft_kernel(const __bf16* __restrict__ Fr, const __bf16* __restrict__ Fi,
                        const __bf16* __restrict__ Xrs, const __bf16* __restrict__ Xis,
                        float* __restrict__ mag, float* __restrict__ out) {
  __shared__ __align__(16) __bf16 sFr[16 * 514];   // 514-bf16 row pitch (TDM pad)
  __shared__ __align__(16) __bf16 sFi[16 * 514];
  const int s     = blockIdx.x;
  const int mtile = blockIdx.y;
  if (threadIdx.x < 32) {
    lfsm_tdm_load_2d((unsigned)(uintptr_t)&sFr[0], Fr + (size_t)mtile * 16 * WW,
                     1u, 512u, 16u, 512u);
    lfsm_tdm_load_2d((unsigned)(uintptr_t)&sFi[0], Fi + (size_t)mtile * 16 * WW,
                     1u, 512u, 16u, 512u);
  }
  __builtin_amdgcn_s_wait_tensorcnt(0);
  __syncthreads();

  const int lane = threadIdx.x & 31;
  const int wv   = threadIdx.x >> 5;
  const int rowbase = (lane & 15) * 514;
  const size_t sb = (size_t)s * SWZT;
  const int b = (s < 16) ? s : (s - 16);
  const size_t N = (size_t)16 * HWF;
  float* phase = out + ((s < 16) ? 4 : 5) * N + (size_t)b * HWF;
  float* magS  = mag + (size_t)s * HWF;

  for (int nt = wv; nt < 17; nt += 8) {
    f32x8 yr = {}, yi = {};
    for (int k0 = 0; k0 < 512; k0 += 32) {
      bf16x16 afr, afi, afin;
#pragma unroll
      for (int e = 0; e < 16; ++e) {
        int li = rowbase + k0 + wmma_a_k(lane, e);
        __bf16 fr = sFr[li];
        float  fi = (float)sFi[li];
        afr[e]  = fr;
        afi[e]  = (__bf16)fi;
        afin[e] = (__bf16)(-fi);
      }
      const size_t boff = (size_t)((k0 >> 5) * 17 + nt) * 512 + lane * 16;
      const bf16x8* bpr = (const bf16x8*)(Xrs + sb + boff);
      const bf16x8* bpi = (const bf16x8*)(Xis + sb + boff);
      __builtin_prefetch(Xrs + sb + boff + 17 * 512, 0, 1);
      bf16x8 r0 = bpr[0], r1 = bpr[1], i0 = bpi[0], i1 = bpi[1];
      bf16x16 bxr = __builtin_shufflevector(r0, r1, 0,1,2,3,4,5,6,7,8,9,10,11,12,13,14,15);
      bf16x16 bxi = __builtin_shufflevector(i0, i1, 0,1,2,3,4,5,6,7,8,9,10,11,12,13,14,15);

      // Yr = Fr@Xr - Fi@Xi ; Yi = Fr@Xi + Fi@Xr
      yr = __builtin_amdgcn_wmma_f32_16x16x32_bf16(false, afr,  false, bxr, (short)0, yr, false, false);
      yr = __builtin_amdgcn_wmma_f32_16x16x32_bf16(false, afin, false, bxi, (short)0, yr, false, false);
      yi = __builtin_amdgcn_wmma_f32_16x16x32_bf16(false, afr,  false, bxi, (short)0, yi, false, false);
      yi = __builtin_amdgcn_wmma_f32_16x16x32_bf16(false, afi,  false, bxr, (short)0, yi, false, false);
    }
    const int ncol = nt * 16 + (lane & 15);
#pragma unroll
    for (int i = 0; i < 8; ++i) {
      int m = mtile * 16 + i + ((lane >> 4) << 3);
      if (ncol < WFQ) {
        float re = yr[i], im = yi[i];
        magS [(size_t)m * WFQ + ncol] = sqrtf(re * re + im * im) + 1e-7f;
        phase[(size_t)m * WFQ + ncol] = atan2f(im, re);
      }
    }
  }
}

// ---------------------------------------------------------------------------
// Fused freq_analyzer conv tower + global pool (per-stream, per 16x16 tile).
// ---------------------------------------------------------------------------
__global__ __launch_bounds__(256)
void lfsm_convpool_kernel(const float* __restrict__ vis, const float* __restrict__ ir,
                          const float* __restrict__ w1, const float* __restrict__ b1,
                          const float* __restrict__ s1, const float* __restrict__ o1,
                          const float* __restrict__ w2, const float* __restrict__ b2,
                          const float* __restrict__ s2, const float* __restrict__ o2,
                          float* __restrict__ pooled) {
  __shared__ float sx[20][20];
  __shared__ float h1[16][18][18];
  __shared__ float sw2[32 * 16 * 9];
  __shared__ float red[32];
  const int s = blockIdx.z;
  const float* img = (s < 16) ? (vis + (size_t)s * HWIMG)
                              : (ir + (size_t)(s - 16) * HWIMG);
  const int tx0 = blockIdx.x * 16, ty0 = blockIdx.y * 16;
  const int tid = threadIdx.x;

  for (int i = tid; i < 32 * 16 * 9; i += 256) sw2[i] = w2[i];
  for (int i = tid; i < 400; i += 256) {
    int ly = i / 20, lx = i % 20;
    int gy = ty0 + ly - 2, gx = tx0 + lx - 2;
    sx[ly][lx] = (gy >= 0 && gy < HH && gx >= 0 && gx < WW) ? img[(size_t)gy * WW + gx] : 0.0f;
  }
  if (tid < 32) red[tid] = 0.0f;
  __syncthreads();

  for (int i = tid; i < 18 * 18; i += 256) {
    int ly = i / 18, lx = i % 18;
    int gy = ty0 + ly - 1, gx = tx0 + lx - 1;
    bool inb = (gy >= 0 && gy < HH && gx >= 0 && gx < WW);
#pragma unroll
    for (int oc = 0; oc < 16; ++oc) {
      float acc = b1[oc];
#pragma unroll
      for (int dy = 0; dy < 3; ++dy)
#pragma unroll
        for (int dx = 0; dx < 3; ++dx)
          acc += sx[ly + dy][lx + dx] * w1[oc * 9 + dy * 3 + dx];
      float v = reluf_(acc * s1[oc] + o1[oc]);
      h1[oc][ly][lx] = inb ? v : 0.0f;
    }
  }
  __syncthreads();

  const int py = tid >> 4, px = tid & 15;
  float acc[32];
#pragma unroll
  for (int oc = 0; oc < 32; ++oc) acc[oc] = b2[oc];
  for (int ic = 0; ic < 16; ++ic) {
    float p00 = h1[ic][py + 0][px + 0], p01 = h1[ic][py + 0][px + 1], p02 = h1[ic][py + 0][px + 2];
    float p10 = h1[ic][py + 1][px + 0], p11 = h1[ic][py + 1][px + 1], p12 = h1[ic][py + 1][px + 2];
    float p20 = h1[ic][py + 2][px + 0], p21 = h1[ic][py + 2][px + 1], p22 = h1[ic][py + 2][px + 2];
    const float* wp = &sw2[ic * 9];
#pragma unroll
    for (int oc = 0; oc < 32; ++oc) {
      const float* w = wp + oc * 144;
      acc[oc] += p00 * w[0] + p01 * w[1] + p02 * w[2]
               + p10 * w[3] + p11 * w[4] + p12 * w[5]
               + p20 * w[6] + p21 * w[7] + p22 * w[8];
    }
  }
#pragma unroll
  for (int oc = 0; oc < 32; ++oc) {
    float v = reluf_(acc[oc] * s2[oc] + o2[oc]);
    atomicAdd(&red[oc], v);
  }
  __syncthreads();
  if (tid < 32) atomicAdd(&pooled[s * 32 + tid], red[tid]);
}

// ---------------------------------------------------------------------------
// Tiny per-sample MLPs: feats -> dynamic kernels + size weights.
// ---------------------------------------------------------------------------
__global__ __launch_bounds__(32)
void lfsm_mlp_kernel(const float* pooled,
                     const float* fa_w3, const float* fa_b3,
                     const float* dk_w1, const float* dk_b1,
                     const float* dk_w2, const float* dk_b2,
                     const float* dk_w3, const float* dk_b3,
                     const float* sw_w1, const float* sw_b1,
                     const float* sw_w2, const float* sw_b2,
                     float* k3, float* k5, float* swv) {
  const int s = threadIdx.x;
  if (s >= NSTR) return;
  const float inv = 1.0f / (float)HWIMG;
  float feats[16];
  for (int j = 0; j < 16; ++j) {
    float a = fa_b3[j];
    for (int i = 0; i < 32; ++i) a += pooled[s * 32 + i] * inv * fa_w3[i * 16 + j];
    feats[j] = a;
  }
  float t1[32];
  for (int j = 0; j < 32; ++j) {
    float a = dk_b1[j];
    for (int i = 0; i < 16; ++i) a += feats[i] * dk_w1[i * 32 + j];
    t1[j] = reluf_(a);
  }
  float t2[64];
  for (int j = 0; j < 64; ++j) {
    float a = dk_b2[j];
    for (int i = 0; i < 32; ++i) a += t1[i] * dk_w2[i * 64 + j];
    t2[j] = reluf_(a);
  }
  for (int j = 0; j < 272; ++j) {
    float a = dk_b3[j];
    for (int i = 0; i < 64; ++i) a += t2[i] * dk_w3[i * 272 + j];
    if (j < 72) k3[s * 72 + j] = a;
    else        k5[s * 200 + (j - 72)] = a;
  }
  float u[32];
  for (int j = 0; j < 32; ++j) {
    float a = sw_b1[j];
    for (int i = 0; i < 16; ++i) a += feats[i] * sw_w1[i * 32 + j];
    u[j] = reluf_(a);
  }
  float l0 = sw_b2[0], l1 = sw_b2[1];
  for (int i = 0; i < 32; ++i) { l0 += u[i] * sw_w2[i * 2 + 0]; l1 += u[i] * sw_w2[i * 2 + 1]; }
  float mx = fmaxf(l0, l1);
  float e0 = expf(l0 - mx), e1 = expf(l1 - mx);
  swv[s * 2 + 0] = e0 / (e0 + e1);
  swv[s * 2 + 1] = e1 / (e0 + e1);
}

// ---------------------------------------------------------------------------
// Shared per-pixel dynamic-conv + freq_enhancer -> e[8]
// ---------------------------------------------------------------------------
__device__ void lfsm_compute_e(const float* __restrict__ mag, int m, int n,
                               const float* ck3, const float* ck5,
                               float sw0, float sw1,
                               const float* fe_w1, const float* fe_b1,
                               const float* fe_s1, const float* fe_o1,
                               const float* fe_w2, const float* fe_b2,
                               float* e_out) {
  float win[5][5];
#pragma unroll
  for (int dy = 0; dy < 5; ++dy)
#pragma unroll
    for (int dx = 0; dx < 5; ++dx) {
      int y = m + dy - 2, x = n + dx - 2;
      win[dy][dx] = (y >= 0 && y < HH && x >= 0 && x < WFQ) ? mag[(size_t)y * WFQ + x] : 0.0f;
    }
  float fu[8];
#pragma unroll
  for (int c = 0; c < 8; ++c) {
    float a3 = 0.0f, a5 = 0.0f;
#pragma unroll
    for (int dy = 0; dy < 3; ++dy)
#pragma unroll
      for (int dx = 0; dx < 3; ++dx)
        a3 += win[dy + 1][dx + 1] * ck3[c * 9 + dy * 3 + dx];
#pragma unroll
    for (int dy = 0; dy < 5; ++dy)
#pragma unroll
      for (int dx = 0; dx < 5; ++dx)
        a5 += win[dy][dx] * ck5[c * 25 + dy * 5 + dx];
    fu[c] = sw0 * a3 + sw1 * a5;
  }
  float t[16];
#pragma unroll
  for (int j = 0; j < 16; ++j) {
    float a = fe_b1[j];
#pragma unroll
    for (int c = 0; c < 8; ++c) a += fu[c] * fe_w1[c * 16 + j];
    t[j] = reluf_(a * fe_s1[j] + fe_o1[j]);
  }
#pragma unroll
  for (int c = 0; c < 8; ++c) {
    float a = fe_b2[c];
#pragma unroll
    for (int j = 0; j < 16; ++j) a += t[j] * fe_w2[j * 8 + c];
    e_out[c] = sigmoidf_(a);
  }
}

// Pass 1: accumulate channel sums of e for squeeze-excite.
__global__ __launch_bounds__(256)
void lfsm_epass1_kernel(const float* __restrict__ mag,
                        const float* __restrict__ k3, const float* __restrict__ k5,
                        const float* __restrict__ swv,
                        const float* fe_w1, const float* fe_b1,
                        const float* fe_s1, const float* fe_o1,
                        const float* fe_w2, const float* fe_b2,
                        float* __restrict__ gsum) {
  __shared__ float ck3[72], ck5[200], red[8];
  const int s = blockIdx.y;
  const int tid = threadIdx.x;
  if (tid < 72)  ck3[tid] = k3[s * 72 + tid];
  if (tid < 200) ck5[tid] = k5[s * 200 + tid];
  if (tid < 8)   red[tid] = 0.0f;
  __syncthreads();
  const int p = blockIdx.x * 256 + tid;
  if (p < HWF) {
    const int m = p / WFQ, n = p % WFQ;
    float e2[8];
    lfsm_compute_e(mag + (size_t)s * HWF, m, n, ck3, ck5,
                   swv[s * 2 + 0], swv[s * 2 + 1],
                   fe_w1, fe_b1, fe_s1, fe_o1, fe_w2, fe_b2, e2);
#pragma unroll
    for (int c = 0; c < 8; ++c) atomicAdd(&red[c], e2[c]);
  }
  __syncthreads();
  if (tid < 8) atomicAdd(&gsum[s * 8 + tid], red[tid]);
}

// Squeeze-excite gate g[s][8]
__global__ __launch_bounds__(32)
void lfsm_gate_kernel(const float* gsum,
                      const float* ff_w1, const float* ff_b1,
                      const float* ff_w2, const float* ff_b2,
                      float* gvec) {
  const int s = threadIdx.x;
  if (s >= NSTR) return;
  const float inv = 1.0f / (float)HWF;
  float gm[8];
  for (int c = 0; c < 8; ++c) gm[c] = gsum[s * 8 + c] * inv;
  float t[16];
  for (int j = 0; j < 16; ++j) {
    float a = ff_b1[j];
    for (int c = 0; c < 8; ++c) a += gm[c] * ff_w1[c * 16 + j];
    t[j] = reluf_(a);
  }
  for (int c = 0; c < 8; ++c) {
    float a = ff_b2[c];
    for (int j = 0; j < 16; ++j) a += t[j] * ff_w2[j * 8 + c];
    gvec[s * 8 + c] = sigmoidf_(a);
  }
}

// Pass 2: recompute e, gate, freq_adapt chain, mask split -> d_out blocks 0..3
__global__ __launch_bounds__(256)
void lfsm_epass2_kernel(const float* __restrict__ mag,
                        const float* __restrict__ k3, const float* __restrict__ k5,
                        const float* __restrict__ swv, const float* __restrict__ gvec,
                        const float* fe_w1, const float* fe_b1,
                        const float* fe_s1, const float* fe_o1,
                        const float* fe_w2, const float* fe_b2,
                        const float* ad_w1, const float* ad_b1,
                        const float* ad_s1, const float* ad_o1,
                        const float* ad_w2, const float* ad_b2,
                        const float* ad_s2, const float* ad_o2,
                        const float* ad_w3, const float* ad_b3,
                        const float* thr_p, const float* scale_p,
                        float* __restrict__ out) {
  __shared__ float ck3[72], ck5[200];
  const int s = blockIdx.y;
  const int tid = threadIdx.x;
  if (tid < 72)  ck3[tid] = k3[s * 72 + tid];
  if (tid < 200) ck5[tid] = k5[s * 200 + tid];
  __syncthreads();
  const int p = blockIdx.x * 256 + tid;
  if (p >= HWF) return;
  const int m = p / WFQ, n = p % WFQ;

  float e2[8];
  lfsm_compute_e(mag + (size_t)s * HWF, m, n, ck3, ck5,
                 swv[s * 2 + 0], swv[s * 2 + 1],
                 fe_w1, fe_b1, fe_s1, fe_o1, fe_w2, fe_b2, e2);
#pragma unroll
  for (int c = 0; c < 8; ++c) e2[c] *= gvec[s * 8 + c];

  float a1[16];
#pragma unroll
  for (int j = 0; j < 16; ++j) {
    float a = ad_b1[j];
#pragma unroll
    for (int c = 0; c < 8; ++c) a += e2[c] * ad_w1[c * 16 + j];
    a1[j] = reluf_(a * ad_s1[j] + ad_o1[j]);
  }
  float a2[8];
#pragma unroll
  for (int c = 0; c < 8; ++c) {
    float a = ad_b2[c];
#pragma unroll
    for (int j = 0; j < 16; ++j) a += a1[j] * ad_w2[j * 8 + c];
    a2[c] = reluf_(a * ad_s2[c] + ad_o2[c]);
  }
  float fa = ad_b3[0];
#pragma unroll
  for (int c = 0; c < 8; ++c) fa += a2[c] * ad_w3[c];
  const float filt = sigmoidf_(fa);
  const float mask = sigmoidf_((filt - thr_p[0]) * scale_p[0]);
  const float hi = fminf(fmaxf(filt * mask, 0.0f), 5.0f);
  const float lo = fminf(fmaxf(filt * (1.0f - mask), 0.0f), 5.0f);

  const int b = (s < 16) ? s : (s - 16);
  const size_t N = (size_t)16 * HWF;
  const size_t pix = (size_t)b * HWF + p;
  if (s < 16) { out[0 * N + pix] = hi; out[1 * N + pix] = lo; }
  else        { out[2 * N + pix] = hi; out[3 * N + pix] = lo; }
}

// ---------------------------------------------------------------------------
// Host launcher
// ---------------------------------------------------------------------------
extern "C" void kernel_launch(void* const* d_in, const int* in_sizes, int n_in,
                              void* d_out, int out_size, void* d_ws, size_t ws_size,
                              hipStream_t stream) {
  (void)in_sizes; (void)n_in; (void)out_size; (void)ws_size;
  const float* vis   = (const float*)d_in[0];
  const float* ir    = (const float*)d_in[1];
  const float* fa_w1 = (const float*)d_in[2];
  const float* fa_b1 = (const float*)d_in[3];
  const float* fa_s1 = (const float*)d_in[4];
  const float* fa_o1 = (const float*)d_in[5];
  const float* fa_w2 = (const float*)d_in[6];
  const float* fa_b2 = (const float*)d_in[7];
  const float* fa_s2 = (const float*)d_in[8];
  const float* fa_o2 = (const float*)d_in[9];
  const float* fa_w3 = (const float*)d_in[10];
  const float* fa_b3 = (const float*)d_in[11];
  const float* dk_w1 = (const float*)d_in[12];
  const float* dk_b1 = (const float*)d_in[13];
  const float* dk_w2 = (const float*)d_in[14];
  const float* dk_b2 = (const float*)d_in[15];
  const float* dk_w3 = (const float*)d_in[16];
  const float* dk_b3 = (const float*)d_in[17];
  const float* sw_w1 = (const float*)d_in[18];
  const float* sw_b1 = (const float*)d_in[19];
  const float* sw_w2 = (const float*)d_in[20];
  const float* sw_b2 = (const float*)d_in[21];
  const float* fe_w1 = (const float*)d_in[22];
  const float* fe_b1 = (const float*)d_in[23];
  const float* fe_s1 = (const float*)d_in[24];
  const float* fe_o1 = (const float*)d_in[25];
  const float* fe_w2 = (const float*)d_in[26];
  const float* fe_b2 = (const float*)d_in[27];
  const float* ad_w1 = (const float*)d_in[28];
  const float* ad_b1 = (const float*)d_in[29];
  const float* ad_s1 = (const float*)d_in[30];
  const float* ad_o1 = (const float*)d_in[31];
  const float* ad_w2 = (const float*)d_in[32];
  const float* ad_b2 = (const float*)d_in[33];
  const float* ad_s2 = (const float*)d_in[34];
  const float* ad_o2 = (const float*)d_in[35];
  const float* ad_w3 = (const float*)d_in[36];
  const float* ad_b3 = (const float*)d_in[37];
  const float* ff_w1 = (const float*)d_in[38];
  const float* ff_b1 = (const float*)d_in[39];
  const float* ff_w2 = (const float*)d_in[40];
  const float* ff_b2 = (const float*)d_in[41];
  const float* thr   = (const float*)d_in[42];
  const float* scale = (const float*)d_in[43];
  float* out = (float*)d_out;

  char* ws = (char*)d_ws;
  size_t off = 0;
  auto take = [&](size_t bytes) -> char* {
    char* p = ws + off;
    off = (off + bytes + 255) & ~(size_t)255;
    return p;
  };
  __bf16* Crs = (__bf16*)take((size_t)SWZT * 2);
  __bf16* Cis = (__bf16*)take((size_t)SWZT * 2);
  __bf16* Fr  = (__bf16*)take((size_t)HWIMG * 2);
  __bf16* Fi  = (__bf16*)take((size_t)HWIMG * 2);
  __bf16* Xrs = (__bf16*)take((size_t)NSTR * SWZT * 2);
  __bf16* Xis = (__bf16*)take((size_t)NSTR * SWZT * 2);
  float* mag    = (float*)take((size_t)NSTR * HWF * 4);
  float* pooled = (float*)take((size_t)NSTR * 32 * 4);
  float* gsum   = (float*)take((size_t)NSTR * 8 * 4);
  float* k3     = (float*)take((size_t)NSTR * 72 * 4);
  float* k5     = (float*)take((size_t)NSTR * 200 * 4);
  float* swv    = (float*)take((size_t)NSTR * 2 * 4);
  float* gvec   = (float*)take((size_t)NSTR * 8 * 4);

  lfsm_zero_kernel<<<dim3(5), 256, 0, stream>>>(pooled, gsum);
  lfsm_twiddle_kernel<<<dim3((HWIMG + 255) / 256), 256, 0, stream>>>(Crs, Cis, Fr, Fi);
  lfsm_convpool_kernel<<<dim3(32, 32, 32), 256, 0, stream>>>(
      vis, ir, fa_w1, fa_b1, fa_s1, fa_o1, fa_w2, fa_b2, fa_s2, fa_o2, pooled);
  lfsm_mlp_kernel<<<1, 32, 0, stream>>>(
      pooled, fa_w3, fa_b3, dk_w1, dk_b1, dk_w2, dk_b2, dk_w3, dk_b3,
      sw_w1, sw_b1, sw_w2, sw_b2, k3, k5, swv);
  lfsm_rowdft_kernel<<<dim3(32, 32), 256, 0, stream>>>(vis, ir, Crs, Cis, Xrs, Xis);
  lfsm_coldft_kernel<<<dim3(32, 32), 256, 0, stream>>>(Fr, Fi, Xrs, Xis, mag, out);
  lfsm_epass1_kernel<<<dim3(HWF / 256, NSTR), 256, 0, stream>>>(
      mag, k3, k5, swv, fe_w1, fe_b1, fe_s1, fe_o1, fe_w2, fe_b2, gsum);
  lfsm_gate_kernel<<<1, 32, 0, stream>>>(gsum, ff_w1, ff_b1, ff_w2, ff_b2, gvec);
  lfsm_epass2_kernel<<<dim3(HWF / 256, NSTR), 256, 0, stream>>>(
      mag, k3, k5, swv, gvec, fe_w1, fe_b1, fe_s1, fe_o1, fe_w2, fe_b2,
      ad_w1, ad_b1, ad_s1, ad_o1, ad_w2, ad_b2, ad_s2, ad_o2, ad_w3, ad_b3,
      thr, scale, out);
}
